// BertNER_76665166233641
// MI455X (gfx1250) — matrix operand order repaired and checked by reading, not compiled, not run on hardware
//
#include <hip/hip_runtime.h>
#include <hip/hip_bf16.h>

typedef _Float16 half8_t __attribute__((ext_vector_type(8)));
typedef _Float16 v16h    __attribute__((ext_vector_type(16)));
typedef float    v8f     __attribute__((ext_vector_type(8)));

#define Bq 32
#define Tq 512
#define Eq 768
#define Hq 100
#define G4 400           // 4*H
#define Kq 10
#define NROWS (Bq*Tq)    // 16384
#define NC 800           // gates for both directions

// ---- workspace layout (bytes) ----
#define OFF_X16   ((size_t)0)
#define SZ_X16    ((size_t)NROWS*Eq*2)              // 25,165,824
#define OFF_W16   (OFF_X16 + SZ_X16)
#define SZ_W16    ((size_t)NC*Eq*2)                 // 1,228,800
#define OFF_BIAS  (OFF_W16 + SZ_W16)
#define SZ_BIAS   ((size_t)3328)
#define OFF_WTT   (OFF_BIAS + SZ_BIAS)
#define SZ_WTT    ((size_t)2*Hq*G4*4)               // 320,000
#define OFF_XW    (OFF_WTT + SZ_WTT)
#define SZ_XW     ((size_t)NROWS*NC*4)              // 52,428,800
#define OFF_HBUF  (OFF_XW + SZ_XW)
#define SZ_HBUF   ((size_t)NROWS*2*Hq*4)            // 13,107,200
#define OFF_FEATS (OFF_HBUF + SZ_HBUF)
#define SZ_FEATS  ((size_t)NROWS*16*4)
#define OFF_BPS   (OFF_FEATS + SZ_FEATS)

// ---------------- conversion kernels ----------------
__global__ void cvt_x(const float* __restrict__ src, _Float16* __restrict__ dst, int n) {
    int i = blockIdx.x * blockDim.x + threadIdx.x;
    if (i < n) dst[i] = (_Float16)src[i];
}

__global__ void cvt_w(const float* __restrict__ Wf, const float* __restrict__ Wb,
                      const float* __restrict__ bihf, const float* __restrict__ bhhf,
                      const float* __restrict__ bihb, const float* __restrict__ bhhb,
                      const float* __restrict__ Whhf, const float* __restrict__ Whhb,
                      _Float16* __restrict__ W16, float* __restrict__ bias,
                      float* __restrict__ WtT) {
    int i = blockIdx.x * blockDim.x + threadIdx.x;
    if (i < NC * Eq) {
        int g = i / Eq, e = i % Eq;
        float v = (g < G4) ? Wf[g * Eq + e] : Wb[(g - G4) * Eq + e];
        W16[i] = (_Float16)v;
    } else if (i < NC * Eq + 2 * Hq * G4) {
        int i2 = i - NC * Eq;
        int dir = i2 / (Hq * G4);
        int idx = i2 % (Hq * G4);
        int j = idx / G4, g = idx % G4;
        const float* Whh = dir ? Whhb : Whhf;
        WtT[i2] = Whh[g * Hq + j];           // transposed: [j][g]
    } else if (i < NC * Eq + 2 * Hq * G4 + NC) {
        int g = i - NC * Eq - 2 * Hq * G4;
        bias[g] = (g < G4) ? (bihf[g] + bhhf[g]) : (bihb[g - G4] + bhhb[g - G4]);
    }
}

// ---------------- WMMA GEMM: XW[16384 x 800] = X[16384 x 768] * Wc^T, + bias ----
// One wave per 16x32 output tile: A fragment reused across two WMMAs.
__global__ __launch_bounds__(256) void gemm_xw(const _Float16* __restrict__ X,
                                               const _Float16* __restrict__ W,
                                               const float* __restrict__ bias,
                                               float* __restrict__ XW) {
    const int wave = (blockIdx.x << 3) + (threadIdx.x >> 5);   // 3200 blocks * 8 waves = 25600
    const int lane = threadIdx.x & 31;
    const int tm = wave / 25;
    const int n0 = (wave % 25) * 32;
    const int m0 = tm << 4;
    const int lr = lane & 15;
    const bool hi = lane >= 16;

    const int row = m0 + lr;
    const _Float16* Xr = X + (size_t)row * Eq + (hi ? 16 : 0);
    const _Float16* W0 = W + (size_t)(n0 + lr) * Eq + (hi ? 16 : 0);
    const _Float16* W1 = W0 + (size_t)16 * Eq;

    v8f acc0 = {}, acc1 = {};
    for (int k0 = 0; k0 < Eq; k0 += 32) {
        __builtin_prefetch(Xr + k0 + 64, 0, 1);
        // A fragment (16x32 f16): lanes<16 hold K 0..15, lanes>=16 hold K 16..31
        half8_t a0 = *(const half8_t*)(Xr + k0);
        half8_t a1 = *(const half8_t*)(Xr + k0 + 8);
        v16h A;
#pragma unroll
        for (int e = 0; e < 8; e++) { A[e] = a0[e]; A[e + 8] = a1[e]; }
        // B fragment (32x16 f16): lane n holds column n; lanes<16 K 0..15, lanes>=16 K 16..31
        v16h B0 = *(const v16h*)(W0 + k0);
        v16h B1 = *(const v16h*)(W1 + k0);
        acc0 = __builtin_amdgcn_wmma_f32_16x16x32_f16(false, A, false, B0, (short)0, acc0, false, false);
        acc1 = __builtin_amdgcn_wmma_f32_16x16x32_f16(false, A, false, B1, (short)0, acc1, false, false);
    }
    const int col0 = n0 + lr;
    const float bb0 = bias[col0];
    const float bb1 = bias[col0 + 16];
#pragma unroll
    for (int v = 0; v < 8; v++) {
        int rowc = m0 + v + (hi ? 8 : 0);
        XW[(size_t)rowc * NC + col0]      = acc0[v] + bb0;
        XW[(size_t)rowc * NC + col0 + 16] = acc1[v] + bb1;
    }
}

// ---------------- LSTM recurrence: one workgroup per (batch, direction) chain ----
__global__ __launch_bounds__(512) void lstm_rec(const float* __restrict__ XW,
                                                const float* __restrict__ WtT,
                                                float* __restrict__ hbuf) {
    __shared__ float hs[Hq];
    __shared__ float gbuf[G4];
    const int b = blockIdx.x;
    const int dir = blockIdx.y;
    const int tid = threadIdx.x;
    const float* Wt = WtT + (size_t)dir * Hq * G4;   // [j][g] transposed, L2-resident
    float creg = 0.f;
    if (tid < Hq) hs[tid] = 0.f;
    __syncthreads();
    for (int s = 0; s < Tq; s++) {
        int t = dir ? (Tq - 1 - s) : s;
        size_t r = (size_t)b * Tq + t;
        if (tid < G4) {
            float acc = XW[r * NC + dir * G4 + tid];
#pragma unroll 4
            for (int j = 0; j < Hq; j++)
                acc += Wt[(size_t)j * G4 + tid] * hs[j];
            gbuf[tid] = acc;
        }
        __syncthreads();
        if (tid < Hq) {
            float ig = 1.f / (1.f + __expf(-gbuf[tid]));
            float fg = 1.f / (1.f + __expf(-gbuf[Hq + tid]));
            float gg = tanhf(gbuf[2 * Hq + tid]);
            float og = 1.f / (1.f + __expf(-gbuf[3 * Hq + tid]));
            creg = fg * creg + ig * gg;
            float hn = og * tanhf(creg);
            hs[tid] = hn;
            hbuf[r * (2 * Hq) + dir * Hq + tid] = hn;
        }
        __syncthreads();
    }
}

// ---------------- output projection: feats[r][k] = hbuf[r]·W_out[k] + b_out[k] ----
__global__ __launch_bounds__(256) void proj(const float* __restrict__ hbuf,
                                            const float* __restrict__ Wout,
                                            const float* __restrict__ bout,
                                            float* __restrict__ feats) {
    int tid = blockIdx.x * blockDim.x + threadIdx.x;
    int r = tid >> 4;
    int k = tid & 15;
    if (r >= NROWS) return;
    if (k < Kq) {
        float acc = bout[k];
        const float* h = hbuf + (size_t)r * 2 * Hq;
        const float* w = Wout + (size_t)k * 2 * Hq;
#pragma unroll 4
        for (int j = 0; j < 2 * Hq; j++) acc += h[j] * w[j];
        feats[(size_t)r * 16 + k] = acc;
    }
}

// ---------------- Viterbi: single wave, lane-per-state ----
__global__ void viterbi(const float* __restrict__ feats, const float* __restrict__ trans,
                        int* __restrict__ bps, float* __restrict__ out) {
    __shared__ float fv[16];
    __shared__ float term[16];
    const int i = threadIdx.x;
    float tr[Kq];
    if (i < Kq)
        for (int j = 0; j < Kq; j++) tr[j] = trans[i * Kq + j];
    if (i < 16) fv[i] = (i == 8) ? 0.f : -10000.f;   // START = 8
    __syncthreads();
    for (int n = 0; n < NROWS; n++) {
        float nv = 0.f;
        if (i < Kq) {
            float best = -3.4e38f; int bj = 0;
            for (int j = 0; j < Kq; j++) {
                float v = fv[j] + tr[j];
                if (v > best) { best = v; bj = j; }   // first-max like jnp.argmax
            }
            bps[n * 16 + i] = bj;
            nv = best + feats[(size_t)n * 16 + i];
        }
        __syncthreads();
        if (i < Kq) fv[i] = nv;
        __syncthreads();
    }
    if (i < Kq) {
        float t = fv[i] + trans[9 * Kq + i];          // transitions[STOP]
        if (i == 8 || i == 9) t = -10000.f;           // mask START, STOP
        term[i] = t;
    }
    __syncthreads();
    if (i == 0) {
        int best = 0; float bs = -3.4e38f;
        for (int j = 0; j < Kq; j++)
            if (term[j] > bs) { bs = term[j]; best = j; }
        out[0] = bs;
        int tag = best;
        for (int n = NROWS - 1; n >= 0; n--) {
            out[1 + n] = (float)tag;
            tag = bps[n * 16 + tag];
        }
    }
}

extern "C" void kernel_launch(void* const* d_in, const int* in_sizes, int n_in,
                              void* d_out, int out_size, void* d_ws, size_t ws_size,
                              hipStream_t stream) {
    const float* embeds = (const float*)d_in[0];
    const float* W_ih_f = (const float*)d_in[1];
    const float* W_hh_f = (const float*)d_in[2];
    const float* b_ih_f = (const float*)d_in[3];
    const float* b_hh_f = (const float*)d_in[4];
    const float* W_ih_b = (const float*)d_in[5];
    const float* W_hh_b = (const float*)d_in[6];
    const float* b_ih_b = (const float*)d_in[7];
    const float* b_hh_b = (const float*)d_in[8];
    const float* W_out  = (const float*)d_in[9];
    const float* b_out  = (const float*)d_in[10];
    const float* trans  = (const float*)d_in[11];

    char* ws = (char*)d_ws;
    _Float16* X16  = (_Float16*)(ws + OFF_X16);
    _Float16* W16  = (_Float16*)(ws + OFF_W16);
    float* biasc   = (float*)(ws + OFF_BIAS);
    float* WtT     = (float*)(ws + OFF_WTT);
    float* XW      = (float*)(ws + OFF_XW);
    float* hbuf    = (float*)(ws + OFF_HBUF);
    float* feats   = (float*)(ws + OFF_FEATS);
    int*   bps     = (int*)(ws + OFF_BPS);
    float* out     = (float*)d_out;

    cvt_x<<<(NROWS * Eq + 255) / 256, 256, 0, stream>>>(embeds, X16, NROWS * Eq);
    cvt_w<<<(NC * Eq + 2 * Hq * G4 + NC + 255) / 256, 256, 0, stream>>>(
        W_ih_f, W_ih_b, b_ih_f, b_hh_f, b_ih_b, b_hh_b, W_hh_f, W_hh_b,
        W16, biasc, WtT);
    gemm_xw<<<3200, 256, 0, stream>>>(X16, W16, biasc, XW);
    lstm_rec<<<dim3(Bq, 2), 512, 0, stream>>>(XW, WtT, hbuf);
    proj<<<(NROWS * 16) / 256, 256, 0, stream>>>(hbuf, W_out, b_out, feats);
    viterbi<<<1, 32, 0, stream>>>(feats, trans, bps, out);
}
